// attention_head_60120952209765
// MI455X (gfx1250) — compile-verified, matrix operand order checked by reading, not run on hardware
//
#include <hip/hip_runtime.h>

#define BT 16384   // B*T
#define T_ 4096
#define E_ 128
#define D_ 64

typedef __attribute__((ext_vector_type(16))) _Float16 v16h;
typedef __attribute__((ext_vector_type(8)))  _Float16 h8;
typedef __attribute__((ext_vector_type(8)))  float    v8f;
typedef __attribute__((ext_vector_type(4)))  float    f4;

static __device__ __forceinline__ v16h cat8(h8 a, h8 b) {
  v16h r;
#pragma unroll
  for (int i = 0; i < 8; ++i) { r[i] = a[i]; r[8 + i] = b[i]; }
  return r;
}

// 16-bit WMMA A/B fragment from row-major halfs:
// element i -> K = (i<8 ? kb8+i : 16+kb8+(i-8)); two aligned 16B chunks.
static __device__ __forceinline__ v16h frag_h(const _Float16* p, int kb8) {
  h8 lo = *(const h8*)(p + kb8);
  h8 hi = *(const h8*)(p + 16 + kb8);
  return cat8(lo, hi);
}

// ------------- prep: pack Wq/Wk/Wv transposed to f16 [64][128] -------------
__global__ __launch_bounds__(256) void pack_w(const float* __restrict__ Wq,
                                              const float* __restrict__ Wk,
                                              const float* __restrict__ Wv,
                                              _Float16* __restrict__ wt) {
  const int idx = blockIdx.x * 256 + threadIdx.x;   // 3 * 64 * 128 = 24576
  const int w = idx >> 13;                          // which matrix
  const int n = (idx >> 7) & 63;                    // output column
  const int k = idx & 127;                          // input row
  const float* W = (w == 0) ? Wq : (w == 1) ? Wk : Wv;
  wt[idx] = (_Float16)W[k * D_ + n];
}

// ---------------- QKV projection: (16384x128) x (128x64), f16 WMMA ---------
__global__ __launch_bounds__(32) void qkv_wmma(const float* __restrict__ x,
                                               const _Float16* __restrict__ wt, // 3x[64][128]
                                               _Float16* __restrict__ qh,   // [BT][64]
                                               _Float16* __restrict__ kh,   // [BT][64]
                                               _Float16* __restrict__ vt)   // [64][BT]
{
  const int lane = threadIdx.x & 31;
  const int l16  = lane & 15;
  const int kb8  = (lane >> 4) * 8;
  const int row0 = blockIdx.x * 16;

  // A fragments of x tile (16 rows x 128): float4 loads, then cvt to f16
  v16h ax[4];
#pragma unroll
  for (int kk = 0; kk < 4; ++kk) {
    const float* base = x + (row0 + l16) * E_ + kk * 32;
    f4 a0 = *(const f4*)(base + kb8);
    f4 a1 = *(const f4*)(base + kb8 + 4);
    f4 b0 = *(const f4*)(base + 16 + kb8);
    f4 b1 = *(const f4*)(base + 16 + kb8 + 4);
#pragma unroll
    for (int i = 0; i < 4; ++i) {
      ax[kk][i]      = (_Float16)a0[i];
      ax[kk][4 + i]  = (_Float16)a1[i];
      ax[kk][8 + i]  = (_Float16)b0[i];
      ax[kk][12 + i] = (_Float16)b1[i];
    }
  }

#pragma unroll
  for (int w = 0; w < 3; ++w) {
    const _Float16* Wt = wt + w * (D_ * E_);        // [64][128]: row n, col k
#pragma unroll
    for (int nt = 0; nt < 4; ++nt) {
      v8f c = {};
#pragma unroll
      for (int kk = 0; kk < 4; ++kk) {
        // B[k][n]: lane holds n = nt*16+l16; contiguous in k -> b128 loads
        v16h bw = frag_h(Wt + (nt * 16 + l16) * E_ + kk * 32, kb8);
        c = __builtin_amdgcn_wmma_f32_16x16x32_f16(false, ax[kk], false, bw,
                                                   (short)0, c, false, false);
      }
      if (w < 2) {
        _Float16* dst = (w == 0) ? qh : kh;
#pragma unroll
        for (int r = 0; r < 8; ++r)      // C layout: row = kb8+r, col = l16
          dst[(row0 + kb8 + r) * D_ + nt * 16 + l16] = (_Float16)c[r];
      } else {
#pragma unroll
        for (int r = 0; r < 8; ++r)      // store V transposed [d][row]
          vt[(nt * 16 + l16) * BT + row0 + kb8 + r] = (_Float16)c[r];
      }
    }
  }
}

// ---------------- Flash attention: 16 queries/wave, 64-key inner blocks ----
__global__ __launch_bounds__(32) void flash_attn(const _Float16* __restrict__ qh,
                                                 const _Float16* __restrict__ kh,
                                                 const _Float16* __restrict__ vt,
                                                 float* __restrict__ out)
{
  const int lane = threadIdx.x & 31;
  const int l16  = lane & 15;
  const int kb8  = (lane >> 4) * 8;

  const int b       = blockIdx.x >> 8;          // 256 query tiles per batch
  const int q0      = (blockIdx.x & 255) * 16;  // local query row base
  const int rowbase = b * T_;

  __shared__ _Float16 lds_p[16][72];            // 64 + 8 pad: 16B-aligned rows

  // Q fragments (16 x 64), 2 d-steps of 32
  v16h aq[2];
#pragma unroll
  for (int s = 0; s < 2; ++s)
    aq[s] = frag_h(qh + (rowbase + q0 + l16) * D_ + s * 32, kb8);

  v8f z = {};
  v8f acc[4];
#pragma unroll
  for (int u = 0; u < 4; ++u) acc[u] = z;
  float mi[8], li[8];
#pragma unroll
  for (int r = 0; r < 8; ++r) { mi[r] = -1e30f; li[r] = 0.f; }

  const int nk = q0 + 16;                       // keys 0 .. q0+15 needed
  for (int j0 = 0; j0 < nk; j0 += 64) {
    const int kleft = nk - j0;                                 // >= 1
    const int nta   = (kleft >= 64) ? 4 : ((kleft + 15) >> 4); // active 16-key tiles
    const bool maskblk = (j0 + 63 > q0);

    // S = Q K^T for keys j0..j0+63: four 16x16 f32 tiles (skip fully-masked)
    v8f st[4];
#pragma unroll
    for (int t = 0; t < 4; ++t) {
      if (t < nta) {
        v8f c = z;
#pragma unroll
        for (int s = 0; s < 2; ++s) {
          v16h bk = frag_h(kh + (rowbase + j0 + t * 16 + l16) * D_ + s * 32, kb8);
          c = __builtin_amdgcn_wmma_f32_16x16x32_f16(false, aq[s], false, bk,
                                                     (short)0, c, false, false);
        }
        st[t] = c;
      } else {
        v8f m;
#pragma unroll
        for (int r = 0; r < 8; ++r) m[r] = -1e30f;
        st[t] = m;
      }
    }

    // online softmax per query row (stats replicated across each 16-lane half)
#pragma unroll
    for (int r = 0; r < 8; ++r) {
      const int qg = q0 + kb8 + r;
      float sv[4];
#pragma unroll
      for (int t = 0; t < 4; ++t) sv[t] = st[t][r] * 0.125f;   // 1/sqrt(64)
      if (maskblk) {
#pragma unroll
        for (int t = 0; t < 4; ++t)
          if (j0 + t * 16 + l16 > qg) sv[t] = -1e30f;
      }
      float mx = fmaxf(fmaxf(sv[0], sv[1]), fmaxf(sv[2], sv[3]));
#pragma unroll
      for (int d = 1; d < 16; d <<= 1) mx = fmaxf(mx, __shfl_xor(mx, d, 32));
      const float mnew  = fmaxf(mi[r], mx);
      const float alpha = __expf(mi[r] - mnew);
      float p[4], rs = 0.f;
#pragma unroll
      for (int t = 0; t < 4; ++t) { p[t] = __expf(sv[t] - mnew); rs += p[t]; }
#pragma unroll
      for (int d = 1; d < 16; d <<= 1) rs += __shfl_xor(rs, d, 32);
      li[r] = li[r] * alpha + rs;
      mi[r] = mnew;
#pragma unroll
      for (int u = 0; u < 4; ++u) acc[u][r] *= alpha;
#pragma unroll
      for (int t = 0; t < 4; ++t)
        lds_p[kb8 + r][t * 16 + l16] = (_Float16)p[t];
    }
    __syncthreads();

    // P in A layout (row = l16), then O += P (16x64) x V (64x64)
    const int nks = (nta + 1) >> 1;             // skip k-step with all-zero P
    const _Float16* pr = &lds_p[l16][0];
#pragma unroll
    for (int s = 0; s < 2; ++s) {
      if (s < nks) {
        v16h ap = cat8(*(const h8*)(pr + s * 32 + kb8),
                       *(const h8*)(pr + s * 32 + 16 + kb8));
#pragma unroll
        for (int u = 0; u < 4; ++u) {
          v16h bv = frag_h(vt + (u * 16 + l16) * BT + rowbase + j0 + s * 32, kb8);
          acc[u] = __builtin_amdgcn_wmma_f32_16x16x32_f16(false, ap, false, bv,
                                                          (short)0, acc[u], false, false);
        }
      }
    }
    __syncthreads();
  }

  // epilogue: divide by running sum, store fp32
#pragma unroll
  for (int r = 0; r < 8; ++r) {
    const float inv = 1.0f / li[r];
#pragma unroll
    for (int u = 0; u < 4; ++u)
      out[(rowbase + q0 + kb8 + r) * D_ + u * 16 + l16] = acc[u][r] * inv;
  }
}

extern "C" void kernel_launch(void* const* d_in, const int* in_sizes, int n_in,
                              void* d_out, int out_size, void* d_ws, size_t ws_size,
                              hipStream_t stream) {
  const float* x  = (const float*)d_in[0];
  const float* Wq = (const float*)d_in[1];
  const float* Wk = (const float*)d_in[2];
  const float* Wv = (const float*)d_in[3];

  _Float16* qh = (_Float16*)d_ws;            // [BT][64]
  _Float16* kh = qh + (size_t)BT * D_;       // [BT][64]
  _Float16* vt = kh + (size_t)BT * D_;       // [64][BT]
  _Float16* wt = vt + (size_t)BT * D_;       // 3 x [64][128] packed f16 weights

  pack_w<<<(3 * D_ * E_) / 256, 256, 0, stream>>>(Wq, Wk, Wv, wt);
  qkv_wmma<<<BT / 16, 32, 0, stream>>>(x, wt, qh, kh, vt);
  flash_attn<<<BT / 16, 32, 0, stream>>>(qh, kh, vt, (float*)d_out);
}
